// MADESplit_49890340110606
// MI455X (gfx1250) — compile-verified, matrix operand order checked by reading, not run on hardware
//
#include <hip/hip_runtime.h>

typedef __bf16 bf16;
typedef __attribute__((ext_vector_type(8)))  __bf16 v8bf;
typedef __attribute__((ext_vector_type(16))) __bf16 v16bf;
typedef __attribute__((ext_vector_type(8)))  float  v8f;

#define MADE_BATCH 4096
#define MADE_DIM   1024   // NUM_INPUTS == NUM_HIDDEN == 1024
#define LDS_LDA    40     // 32 K-elems + 8 pad  -> conflict-free ds_load_b128

enum { ACT_NONE = 0, ACT_TANH = 1, ACT_RELU = 2 };

// ---------------------------------------------------------------------------
// One-time prep: masked weights -> bf16 (6 matrices, each 1024x1024).
// Order in workspace: s_w1, s_w2, s_w3, t_w1, t_w2, t_w3
// ---------------------------------------------------------------------------
__global__ __launch_bounds__(256) void prep_weights(
    const float* __restrict__ w0, const float* __restrict__ w1,
    const float* __restrict__ w2, const float* __restrict__ w3,
    const float* __restrict__ w4, const float* __restrict__ w5,
    const float* __restrict__ m_in, const float* __restrict__ m_hid,
    const float* __restrict__ m_out, bf16* __restrict__ out)
{
    size_t i   = (size_t)blockIdx.x * blockDim.x + threadIdx.x;
    int    mat = (int)(i >> 20);          // which 1M-element matrix
    size_t off = i & 0xFFFFFu;
    const float* w; const float* msk;
    switch (mat) {
        case 0:  w = w0; msk = m_in;  break;
        case 1:  w = w1; msk = m_hid; break;
        case 2:  w = w2; msk = m_out; break;
        case 3:  w = w3; msk = m_in;  break;
        case 4:  w = w4; msk = m_hid; break;
        default: w = w5; msk = m_out; break;
    }
    out[i] = (bf16)(w[off] * msk[off]);
}

__global__ __launch_bounds__(256) void cvt_bf16(
    const float* __restrict__ x, bf16* __restrict__ y)
{
    size_t i = (size_t)blockIdx.x * blockDim.x + threadIdx.x;
    y[i] = (bf16)x[i];
}

// ---------------------------------------------------------------------------
// LDS fragment loader: 16 bf16 = two 16-byte chunks 16 elems apart
// (CDNA5 16-bit fragment layout: V0-3 <- K[base..base+7], V4-7 <- K[base+16..23];
//  the +8 half-wave shift is handled by the caller via `half*8`).
// ---------------------------------------------------------------------------
__device__ __forceinline__ v16bf lds_frag16(const bf16* p)
{
    v8bf lo = *reinterpret_cast<const v8bf*>(p);
    v8bf hi = *reinterpret_cast<const v8bf*>(p + 16);
    return __builtin_shufflevector(lo, hi,
        0, 1, 2, 3, 4, 5, 6, 7, 8, 9, 10, 11, 12, 13, 14, 15);
}

// ---------------------------------------------------------------------------
// Async-stage one 128x32 bf16 slab (global, row-major lda=1024) into LDS
// (padded lda = LDS_LDA). 256 threads x 2 chunks x 16B = 8 KB.
// Tracked by ASYNCcnt; caller does s_wait_asynccnt + barrier.
// ---------------------------------------------------------------------------
__device__ __forceinline__ void stage_async(
    const bf16* __restrict__ gbase, int rowBlock, int k,
    bf16* lbuf, int tid)
{
#pragma unroll
    for (int c = 0; c < 2; ++c) {
        const int chunk = tid + c * 256;      // 0..511
        const int row   = chunk >> 2;         // 0..127
        const int kc    = (chunk & 3) * 8;    // 0,8,16,24
        const bf16* g = gbase + (size_t)(rowBlock + row) * MADE_DIM + k + kc;
        const unsigned l = (unsigned)(uintptr_t)(lbuf + row * LDS_LDA + kc);
        asm volatile("global_load_async_to_lds_b128 %0, %1, off"
                     :: "v"(l), "v"(g) : "memory");
    }
}

__device__ __forceinline__ void wait_async_all()
{
    asm volatile("s_wait_asynccnt 0x0" ::: "memory");
}

// ---------------------------------------------------------------------------
// WMMA GEMM:  C[M=4096, N=1024] = act(A[M,K=1024] @ W[N,K]^T + bias)
//   - 256 threads = 8 wave32 waves arranged 4(M) x 2(N)
//   - wave tile 32x64 = 2x4 V_WMMA_F32_16X16X32_BF16 tiles (64 acc VGPRs)
//   - block tile 128(M) x 128(N); grid = (N/128, M/128) = (8, 32)
//   - A/B slabs double-buffered in LDS via global_load_async_to_lds_b128
// ---------------------------------------------------------------------------
template <int ACT, bool OUTBF>
__global__ __launch_bounds__(256) void gemm_wmma(
    const bf16* __restrict__ A, const bf16* __restrict__ W,
    const float* __restrict__ bias,
    float* __restrict__ Cf, bf16* __restrict__ Cb)
{
    constexpr int K = MADE_DIM;
    constexpr int N = MADE_DIM;

    __shared__ bf16 sA[2][128 * LDS_LDA];   // 2 x 10 KB
    __shared__ bf16 sB[2][128 * LDS_LDA];   // 2 x 10 KB

    const int tid  = threadIdx.x;
    const int lane = tid & 31;
    const int wave = tid >> 5;
    const int l16  = lane & 15;
    const int half = lane >> 4;             // 0: lanes 0-15, 1: lanes 16-31
    const int wm   = wave >> 1;             // 0..3 -> M tile
    const int wn   = wave & 1;              // 0..1 -> N tile

    const int mBlock = blockIdx.y * 128;
    const int nBlock = blockIdx.x * 128;

    v8f acc[2][4] = {};

    // prologue: stage k=0 into buffer 0
    stage_async(A, mBlock, 0, sA[0], tid);
    stage_async(W, nBlock, 0, sB[0], tid);
    wait_async_all();
    __syncthreads();

    for (int k = 0; k < K; k += 32) {
        const int cur = (k >> 5) & 1;

        // stage next slab into the other buffer while computing
        if (k + 32 < K) {
            stage_async(A, mBlock, k + 32, sA[cur ^ 1], tid);
            stage_async(W, nBlock, k + 32, sB[cur ^ 1], tid);
        }

        // fragment reads from LDS (conflict-free: row stride 20 banks)
        v16bf af[2], bf[4];
#pragma unroll
        for (int i = 0; i < 2; ++i)
            af[i] = lds_frag16(&sA[cur][(wm * 32 + i * 16 + l16) * LDS_LDA + half * 8]);
#pragma unroll
        for (int j = 0; j < 4; ++j)
            bf[j] = lds_frag16(&sB[cur][(wn * 64 + j * 16 + l16) * LDS_LDA + half * 8]);

#pragma unroll
        for (int i = 0; i < 2; ++i)
#pragma unroll
            for (int j = 0; j < 4; ++j)
                acc[i][j] = __builtin_amdgcn_wmma_f32_16x16x32_bf16(
                    false, af[i], false, bf[j], (short)0, acc[i][j], false, false);

        wait_async_all();
        __syncthreads();
    }

    // C/D layout: VGPR v, lanes 0-15 -> (M=v, N=lane); lanes 16-31 -> (M=v+8)
#pragma unroll
    for (int i = 0; i < 2; ++i) {
#pragma unroll
        for (int j = 0; j < 4; ++j) {
            const int   n  = nBlock + wn * 64 + j * 16 + l16;
            const float bv = bias[n];
#pragma unroll
            for (int v = 0; v < 8; ++v) {
                const int m = mBlock + wm * 32 + i * 16 + half * 8 + v;
                float val = acc[i][j][v] + bv;
                if constexpr (ACT == ACT_TANH) val = tanhf(val);
                if constexpr (ACT == ACT_RELU) val = fmaxf(val, 0.0f);
                if constexpr (OUTBF) Cb[(size_t)m * N + n] = (bf16)val;
                else                 Cf[(size_t)m * N + n] = val;
            }
        }
    }
}

// ---------------------------------------------------------------------------
// u = (x - m) * exp(-a)   (m pre-staged in u region of d_out, updated in place)
// logdet[row] = -sum_c a[row, c]
// ---------------------------------------------------------------------------
__global__ __launch_bounds__(256) void finalize(
    const float* __restrict__ x, const float* __restrict__ a,
    float* __restrict__ u, float* __restrict__ logdet)
{
    __shared__ float red[256];
    const int    row  = blockIdx.x;
    const size_t base = (size_t)row * MADE_DIM;

    float s = 0.0f;
    for (int c = threadIdx.x; c < MADE_DIM; c += 256) {
        const float av = a[base + c];
        const float mv = u[base + c];          // m was written here
        u[base + c] = (x[base + c] - mv) * __expf(-av);
        s += av;
    }
    red[threadIdx.x] = s;
    __syncthreads();
    for (int off = 128; off > 0; off >>= 1) {
        if ((int)threadIdx.x < off) red[threadIdx.x] += red[threadIdx.x + off];
        __syncthreads();
    }
    if (threadIdx.x == 0) logdet[row] = -red[0];
}

// ---------------------------------------------------------------------------
extern "C" void kernel_launch(void* const* d_in, const int* in_sizes, int n_in,
                              void* d_out, int out_size, void* d_ws, size_t ws_size,
                              hipStream_t stream)
{
    (void)in_sizes; (void)n_in; (void)out_size; (void)ws_size;

    const float* x       = (const float*)d_in[0];
    const float* s_w1    = (const float*)d_in[1];
    const float* s_b1    = (const float*)d_in[2];
    const float* s_w2    = (const float*)d_in[3];
    const float* s_b2    = (const float*)d_in[4];
    const float* s_w3    = (const float*)d_in[5];
    const float* s_b3    = (const float*)d_in[6];
    const float* t_w1    = (const float*)d_in[7];
    const float* t_b1    = (const float*)d_in[8];
    const float* t_w2    = (const float*)d_in[9];
    const float* t_b2    = (const float*)d_in[10];
    const float* t_w3    = (const float*)d_in[11];
    const float* t_b3    = (const float*)d_in[12];
    const float* mask_in  = (const float*)d_in[13];
    const float* mask_hid = (const float*)d_in[14];
    const float* mask_out = (const float*)d_in[15];

    const size_t WSZ = (size_t)MADE_DIM * MADE_DIM;        // 1M elems per matrix
    const size_t XSZ = (size_t)MADE_BATCH * MADE_DIM;      // 4M elems

    char*  ws  = (char*)d_ws;
    bf16*  Wb  = (bf16*)ws;                                 // 6 * 2 MB
    bf16*  xb  = (bf16*)(ws + 6 * WSZ * sizeof(bf16));      // 8 MB
    bf16*  h1  = (bf16*)(ws + 6 * WSZ * sizeof(bf16) + 1 * XSZ * sizeof(bf16));
    bf16*  h2  = (bf16*)(ws + 6 * WSZ * sizeof(bf16) + 2 * XSZ * sizeof(bf16));
    float* a_f = (float*)(ws + 6 * WSZ * sizeof(bf16) + 3 * XSZ * sizeof(bf16));

    float* u      = (float*)d_out;
    float* logdet = u + XSZ;

    prep_weights<<<(unsigned)(6 * WSZ / 256), 256, 0, stream>>>(
        s_w1, s_w2, s_w3, t_w1, t_w2, t_w3, mask_in, mask_hid, mask_out, Wb);
    cvt_bf16<<<(unsigned)(XSZ / 256), 256, 0, stream>>>(x, xb);

    dim3 grid(MADE_DIM / 128, MADE_BATCH / 128);   // (8, 32)

    // s branch: tanh, tanh, none -> m (staged into u region of d_out)
    gemm_wmma<ACT_TANH, true ><<<grid, 256, 0, stream>>>(xb, Wb + 0 * WSZ, s_b1, nullptr, h1);
    gemm_wmma<ACT_TANH, true ><<<grid, 256, 0, stream>>>(h1, Wb + 1 * WSZ, s_b2, nullptr, h2);
    gemm_wmma<ACT_NONE, false><<<grid, 256, 0, stream>>>(h2, Wb + 2 * WSZ, s_b3, u, nullptr);

    // t branch: relu, relu, none -> a (workspace)
    gemm_wmma<ACT_RELU, true ><<<grid, 256, 0, stream>>>(xb, Wb + 3 * WSZ, t_b1, nullptr, h1);
    gemm_wmma<ACT_RELU, true ><<<grid, 256, 0, stream>>>(h1, Wb + 4 * WSZ, t_b2, nullptr, h2);
    gemm_wmma<ACT_NONE, false><<<grid, 256, 0, stream>>>(h2, Wb + 5 * WSZ, t_b3, a_f, nullptr);

    finalize<<<MADE_BATCH, 256, 0, stream>>>(x, a_f, u, logdet);
}
